// GlobalAttentionPool_16312285790334
// MI455X (gfx1250) — matrix operand order repaired and checked by reading, not compiled, not run on hardware
//
#include <hip/hip_runtime.h>
#include <hip/hip_bf16.h>

typedef __bf16 bf16_t;
typedef __attribute__((ext_vector_type(16))) __bf16 v16bf;
typedef __attribute__((ext_vector_type(8)))  __bf16 v8bf;
typedef __attribute__((ext_vector_type(8)))  float  v8f;
typedef unsigned int u32x4 __attribute__((ext_vector_type(4)));
typedef int          i32x4 __attribute__((ext_vector_type(4)));
typedef int          i32x8 __attribute__((ext_vector_type(8)));

#define NPTS   262144
#define IN_DIM 256
#define HIDDEN 256
#define NSEG   1024
#define MTILE  128   // rows of x per block; 8 waves x 16-row wave tiles

#if defined(__has_builtin)
#if __has_builtin(__builtin_amdgcn_tensor_load_to_lds) && __has_builtin(__builtin_amdgcn_s_wait_tensorcnt)
#define HAVE_TDM 1
#endif
#endif

__device__ __forceinline__ bf16_t f2bf(float f) {
  unsigned u = __builtin_bit_cast(unsigned, f);
  unsigned r = u + 0x7FFFu + ((u >> 16) & 1u);   // round-to-nearest-even
  return __builtin_bit_cast(bf16_t, (unsigned short)(r >> 16));
}

// ---------------- Kernel 0: W1 (f32, [K][N]) -> W1T bf16 [N][K] ----------------
__global__ void convert_w1_kernel(const float* __restrict__ W1,
                                  bf16_t* __restrict__ W1T) {
  int i = blockIdx.x * blockDim.x + threadIdx.x;   // 0..65535
  int k = i / HIDDEN;
  int n = i % HIDDEN;
  W1T[n * IN_DIM + k] = f2bf(W1[i]);
}

// ---------------- Kernel 0b: zero segment scratch ----------------
__global__ void zero_seg_kernel(int* __restrict__ segmax_i,
                                float* __restrict__ segsum) {
  int i = blockIdx.x * blockDim.x + threadIdx.x;
  if (i < NSEG) { segmax_i[i] = 0; segsum[i] = 0.0f; }
}

#ifdef HAVE_TDM
// TDM: DMA 256x256 bf16 tensor (row-major contiguous, tile == tensor) into LDS.
__device__ __forceinline__ void tdm_load_w1(const bf16_t* gsrc, bf16_t* ldst) {
  const unsigned lds_off = (unsigned)(unsigned long long)ldst;   // addr[31:0] = LDS offset
  const unsigned long long ga = (unsigned long long)gsrc;

  u32x4 g0;
  g0[0] = 1u;                                            // count=1, no gather/restore
  g0[1] = lds_off;                                       // lds_addr
  g0[2] = (unsigned)(ga & 0xFFFFFFFFu);                  // global_addr[31:0]
  g0[3] = (unsigned)((ga >> 32) & 0x01FFFFFFu) | (2u << 30); // global_addr[56:32] | type=2

  i32x8 g1;
  g1[0] = 0x00010000;            // wg_mask=0, data_size=1 (2 bytes)
  g1[1] = (int)(256u << 16);     // tensor_dim0 = 256 (bits 79:48)
  g1[2] = (int)(256u << 16);     // tensor_dim0 hi=0 ; tensor_dim1 = 256 (bits 111:80)
  g1[3] = (int)(256u << 16);     // tensor_dim1 hi=0 ; tile_dim0 = 256 (bits 127:112)
  g1[4] = 256;                   // tile_dim1 = 256 ; tile_dim2 = 0
  g1[5] = 256;                   // tensor_dim0_stride = 256 (data_size units)
  g1[6] = 0;                     // stride hi ; tensor_dim1_stride = 0
  g1[7] = 0;

  i32x4 z4 = {0, 0, 0, 0};
#if __clang_major__ >= 23
  i32x8 z8 = {0, 0, 0, 0, 0, 0, 0, 0};
  __builtin_amdgcn_tensor_load_to_lds(g0, g1, z4, z4, z8, 0);
#else
  __builtin_amdgcn_tensor_load_to_lds(g0, g1, z4, z4, 0);
#endif
}
#endif

// ---------------- Kernel 1: fused gate MLP -> scores, via bf16 WMMA ----------------
__global__ void score_kernel(const float* __restrict__ x,
                             const bf16_t* __restrict__ W1T,
                             const float* __restrict__ b1,
                             const float* __restrict__ W2,
                             const float* __restrict__ b2,
                             float* __restrict__ scores) {
  extern __shared__ char smem[];
  bf16_t* sW = (bf16_t*)smem;                                      // [HIDDEN][IN_DIM] 128KB
  bf16_t* sX = (bf16_t*)(smem + IN_DIM * HIDDEN * sizeof(bf16_t)); // [MTILE][IN_DIM]   64KB

  const int t    = threadIdx.x;
  const int lane = t & 31;
  const int wave = t >> 5;
  const int rowBase = blockIdx.x * MTILE;

#ifdef HAVE_TDM
  // Wave 0 kicks off the TDM DMA of W1T into LDS; other waves skip (TDM ignores
  // EXEC, so a wave-uniform branch avoids duplicate descriptors).
  if (t < 32) tdm_load_w1(W1T, sW);
#else
  {
    const uint4* src = (const uint4*)W1T;
    uint4*       dst = (uint4*)sW;
    #pragma unroll
    for (int i = 0; i < (IN_DIM * HIDDEN * 2 / 16) / 256; ++i)
      dst[t + i * 256] = src[t + i * 256];
  }
#endif

  // All waves convert the x tile f32 -> bf16 (overlaps the TDM transfer).
  {
    const float4* src = (const float4*)(x + (size_t)rowBase * IN_DIM);
    #pragma unroll
    for (int i = 0; i < (MTILE * IN_DIM / 4) / 256; ++i) {         // 32 iters
      float4 v = src[t + i * 256];
      int base = (t + i * 256) * 4;
      sX[base + 0] = f2bf(v.x);
      sX[base + 1] = f2bf(v.y);
      sX[base + 2] = f2bf(v.z);
      sX[base + 3] = f2bf(v.w);
    }
  }
#ifdef HAVE_TDM
  __builtin_amdgcn_s_wait_tensorcnt(0);
#endif
  __syncthreads();

  const int am    = lane & 15;            // A: row within wave tile
  const int akoff = (lane >> 4) * 16;     // A: lanes 0-15 K=0..15; 16-31 K=16..31
  const int bn    = lane & 15;            // B: column
  const int bkoff = (lane >> 4) * 16;     // B: lanes 0-15 K=0..15; 16-31 K=16..31

  // Preload all 8 A fragments for this wave's 16 rows (64 VGPRs, reused 16x).
  const bf16_t* aRow = sX + (wave * 16 + am) * IN_DIM;
  v16bf afrag[8];
  #pragma unroll
  for (int kt = 0; kt < 8; ++kt) {
    const v8bf* ap = (const v8bf*)(aRow + kt * 32 + akoff);
    afrag[kt] = __builtin_shufflevector(ap[0], ap[1],
        0, 1, 2, 3, 4, 5, 6, 7, 8, 9, 10, 11, 12, 13, 14, 15);
  }

  // N-tile outer loop: one 8-VGPR accumulator live at a time (no spills).
  float part[8];
  #pragma unroll
  for (int r = 0; r < 8; ++r) part[r] = 0.0f;

  const int n0 = lane & 15;
  #pragma unroll
  for (int nt = 0; nt < 16; ++nt) {
    v8f acc = {};
    const bf16_t* bRow = sW + (nt * 16 + bn) * IN_DIM + bkoff;
    #pragma unroll
    for (int kt = 0; kt < 8; ++kt) {
      const v16bf bfrag = *(const v16bf*)(bRow + kt * 32);
      acc = __builtin_amdgcn_wmma_f32_16x16x32_bf16(
          false, afrag[kt], false, bfrag, (short)0, acc, false, false);
    }
    // Fold this N-tile into the per-lane score partials: h=ReLU(acc+b1[n]); +h*W2[n].
    const int n = nt * 16 + n0;
    const float bb  = b1[n];
    const float w2v = W2[n];
    #pragma unroll
    for (int r = 0; r < 8; ++r) {
      float h = acc[r] + bb;
      h = h > 0.0f ? h : 0.0f;
      part[r] += h * w2v;
    }
  }

  // Reduce over the 16 lanes of each half (xor masks 1..8 stay within halves).
  #pragma unroll
  for (int m = 8; m >= 1; m >>= 1) {
    #pragma unroll
    for (int r = 0; r < 8; ++r)
      part[r] += __shfl_xor(part[r], m, 32);
  }

  if ((lane & 15) == 0) {
    const int row = rowBase + wave * 16 + ((lane >> 4) ? 8 : 0);
    const float bias2 = b2[0];
    #pragma unroll
    for (int r = 0; r < 8; ++r)
      scores[row + r] = part[r] + bias2;
  }
}

// ---------------- Kernel 2a: segment max (int-bits atomicMax; init 0 == clamp >= 0) ----
__global__ void segmax_kernel(const float* __restrict__ scores,
                              const int* __restrict__ batch,
                              int* __restrict__ segmax_i) {
  int i = blockIdx.x * blockDim.x + threadIdx.x;
  if (i < NPTS)
    atomicMax(&segmax_i[batch[i]], __float_as_int(scores[i]));
}

// ---------------- Kernel 2b: w = exp(score - segmax); segment sum ----------------
__global__ void segsum_kernel(const float* __restrict__ scores,
                              const int* __restrict__ batch,
                              const int* __restrict__ segmax_i,
                              float* __restrict__ wbuf,
                              float* __restrict__ segsum) {
  int i = blockIdx.x * blockDim.x + threadIdx.x;
  if (i < NPTS) {
    int g = batch[i];
    float m = __int_as_float(segmax_i[g]);
    float e = __expf(scores[i] - m);
    wbuf[i] = e;
    atomicAdd(&segsum[g], e);
  }
}

// ---------------- Kernel 3: out[g,:] = sum_{n in g} x[n,:] * w[n] / (segsum+eps) -------
__global__ void pool_kernel(const float* __restrict__ x,
                            const int* __restrict__ batch,
                            const float* __restrict__ wbuf,
                            const float* __restrict__ segsum,
                            float* __restrict__ out) {
  const int g = blockIdx.x;
  const int d = threadIdx.x;   // 256 threads == IN_DIM

  // lower_bound(batch, g) and lower_bound(batch, g+1) on the sorted batch ids.
  int lo = 0, hi = NPTS;
  while (lo < hi) { int mid = (lo + hi) >> 1; if (batch[mid] < g) lo = mid + 1; else hi = mid; }
  const int start = lo;
  hi = NPTS;
  while (lo < hi) { int mid = (lo + hi) >> 1; if (batch[mid] < g + 1) lo = mid + 1; else hi = mid; }
  const int end = lo;

  const float inv = 1.0f / (segsum[g] + 1e-9f);
  float acc = 0.0f;
  for (int n = start; n < end; ++n) {
    if (n + 4 < end)
      __builtin_prefetch(&x[(size_t)(n + 4) * IN_DIM + d], 0, 0);
    acc += x[(size_t)n * IN_DIM + d] * wbuf[n];
  }
  out[g * IN_DIM + d] = acc * inv;
}

// ---------------- Host-side launcher ----------------
extern "C" void kernel_launch(void* const* d_in, const int* in_sizes, int n_in,
                              void* d_out, int out_size, void* d_ws, size_t ws_size,
                              hipStream_t stream) {
  (void)in_sizes; (void)n_in; (void)out_size; (void)ws_size;

  const float* x     = (const float*)d_in[0];
  const float* W1    = (const float*)d_in[1];
  const float* b1    = (const float*)d_in[2];
  const float* W2    = (const float*)d_in[3];
  const float* b2    = (const float*)d_in[4];
  const int*   batch = (const int*)d_in[5];
  float*       out   = (float*)d_out;

  char* ws = (char*)d_ws;
  bf16_t* W1T    = (bf16_t*)ws;                                        // 128 KB
  float*  scores = (float*)(ws + 128 * 1024);                          // 1 MB
  float*  wbuf   = (float*)(ws + 128 * 1024 + NPTS * 4);               // 1 MB
  int*    segmax = (int*)  (ws + 128 * 1024 + 2 * NPTS * 4);           // 4 KB
  float*  segsum = (float*)(ws + 128 * 1024 + 2 * NPTS * 4 + NSEG * 4);// 4 KB

  convert_w1_kernel<<<(IN_DIM * HIDDEN) / 256, 256, 0, stream>>>(W1, W1T);
  zero_seg_kernel<<<(NSEG + 255) / 256, 256, 0, stream>>>(segmax, segsum);

  const size_t ldsBytes = (size_t)(IN_DIM * HIDDEN + MTILE * IN_DIM) * sizeof(bf16_t); // 192 KB
  score_kernel<<<NPTS / MTILE, 256, ldsBytes, stream>>>(x, W1T, b1, W2, b2, scores);

  segmax_kernel<<<NPTS / 256, 256, 0, stream>>>(scores, batch, segmax);
  segsum_kernel<<<NPTS / 256, 256, 0, stream>>>(scores, batch, segmax, wbuf, segsum);
  pool_kernel<<<NSEG, IN_DIM, 0, stream>>>(x, batch, wbuf, segsum, out);
}
